// KDE_gau_64690797412758
// MI455X (gfx1250) — compile-verified, matrix operand order checked by reading, not run on hardware
//
#include <hip/hip_runtime.h>

typedef __attribute__((ext_vector_type(16))) _Float16 v16h;
typedef __attribute__((ext_vector_type(8)))  float    v8f;
typedef __attribute__((ext_vector_type(4)))  float    v4f;

typedef __attribute__((address_space(1))) int gas_int;   // global (AS1)
typedef __attribute__((address_space(3))) int las_int;   // LDS (AS3)

#define LEN            32768
#define NB             16
#define NK             256
#define WG_THREADS     512           // 16 waves -> 16 row-tiles of 16 grid points
#define CHUNK          512           // samples staged in LDS per round (2 KB)
#define SAMPLES_PER_WG 2048
#define SPLITS         (LEN / SAMPLES_PER_WG)   // 16 -> 256 workgroups total
#define NCHUNK         (SAMPLES_PER_WG / CHUNK) // 4

// -0.5 * (1/h)^2 * log2(e), h = 0.1  =>  exp(-0.5*z^2) = exp2(d*d*KEXP)
#define KEXP  (-72.13475204444817f)
// (1/h) * 1/sqrt(2*pi) / LEN
#define SCALE (10.0f * 0.3989422804014327f / (float)LEN)

#if __has_builtin(__builtin_amdgcn_global_load_async_to_lds_b32)
#define HAVE_ASYNC_LDS 1
#else
#define HAVE_ASYNC_LDS 0
#endif

__global__ __launch_bounds__(256) void kde_zero(float* out) {
    out[blockIdx.x * 256 + threadIdx.x] = 0.0f;
}

// Stage one f32 per thread from global into LDS.
// Async path: global_load_async_to_lds_b32 (no VGPR round-trip, ASYNCcnt).
__device__ __forceinline__ void stage(const float* g, float* l) {
#if HAVE_ASYNC_LDS
    __builtin_amdgcn_global_load_async_to_lds_b32(
        (gas_int*)const_cast<float*>(g),
        (las_int*)l,
        /*offset=*/0, /*cpol=*/0);
#else
    __builtin_prefetch(g + CHUNK, 0, 0);   // global_prefetch_b8 for next chunk
    *l = *g;
#endif
}

__device__ __forceinline__ void stage_wait() {
#if HAVE_ASYNC_LDS
#if __has_builtin(__builtin_amdgcn_s_wait_asynccnt)
    __builtin_amdgcn_s_wait_asynccnt(0);
#else
    asm volatile("s_wait_asynccnt 0x0" ::: "memory");
#endif
#endif
}

__global__ __launch_bounds__(WG_THREADS) void kde_wmma(const float* __restrict__ data,
                                                       const float* __restrict__ cX,
                                                       float* __restrict__ out) {
    __shared__ float xs[2][CHUNK];       // ping-pong staging buffers (4 KB)

    const int b    = blockIdx.x / SPLITS;
    const int s0   = (blockIdx.x % SPLITS) * SAMPLES_PER_WG;
    const int tid  = threadIdx.x;
    const int wave = tid >> 5;
    const int lane = tid & 31;
    const int m    = lane & 15;          // matrix row within this wave's tile
    const int row0 = wave * 16;          // tile base grid-point index

    const float c = cX[row0 + m];        // each lane's grid point

    v8f acc = {};                        // f32 accumulator (C/D of WMMA)
    v16h ones;
#pragma unroll
    for (int j = 0; j < 16; ++j) ones[j] = (_Float16)1.0f;

    const float* src = data + (size_t)b * LEN + s0;

    // Prologue: stage chunk 0, wait, sync.
    stage(src + tid, &xs[0][tid]);
    stage_wait();
    __syncthreads();

    for (int i = 0; i < NCHUNK; ++i) {
        const int cur = i & 1;

        // Kick off the async DMA of the next chunk into the other buffer;
        // it overlaps with the 16 WMMA steps below.
        if (i + 1 < NCHUNK)
            stage(src + (i + 1) * CHUNK + tid, &xs[cur ^ 1][tid]);

        // 16 WMMA steps of 32 samples each. Lane packs 16 f16 exp terms for
        // its row; since B == ones, K-slot assignment is free: low half-wave
        // takes samples [0,16), high half-wave takes [16,32) of the step.
#pragma unroll 4
        for (int t = 0; t < CHUNK / 32; ++t) {
            const float* xb = &xs[cur][t * 32 + ((lane >> 4) << 4)];
            v4f x0 = *(const v4f*)(xb + 0);
            v4f x1 = *(const v4f*)(xb + 4);
            v4f x2 = *(const v4f*)(xb + 8);
            v4f x3 = *(const v4f*)(xb + 12);

            v16h a;
#pragma unroll
            for (int j = 0; j < 4; ++j) {
                float d0 = x0[j] - c;
                float d1 = x1[j] - c;
                float d2 = x2[j] - c;
                float d3 = x3[j] - c;
                a[j]      = (_Float16)__builtin_amdgcn_exp2f(d0 * d0 * KEXP);
                a[4 + j]  = (_Float16)__builtin_amdgcn_exp2f(d1 * d1 * KEXP);
                a[8 + j]  = (_Float16)__builtin_amdgcn_exp2f(d2 * d2 * KEXP);
                a[12 + j] = (_Float16)__builtin_amdgcn_exp2f(d3 * d3 * KEXP);
            }
            // D = A(16x32 f16 exp terms) x ones(32x16) + C -> row sums in every column
            acc = __builtin_amdgcn_wmma_f32_16x16x32_f16(
                /*neg_a=*/false, a, /*neg_b=*/false, ones,
                /*c_mod=*/(short)0, acc, /*reuse_a=*/false, /*reuse_b=*/false);
        }

        // Drain this wave's async DMA, then make the freshly-written buffer
        // (and the completed reads of the current one) visible to all waves.
        if (i + 1 < NCHUNK) {
            stage_wait();
            __syncthreads();
        }
    }

    // Column N=0 of the 16x16 f32 C/D tile lives in lane 0 (rows M=0..7 in
    // VGPRs 0..7) and lane 16 (rows M=8..15). Scale and atomically merge.
    if ((lane & 15) == 0) {
        const int mbase = row0 + ((lane >> 4) << 3);
#pragma unroll
        for (int j = 0; j < 8; ++j)
            atomicAdd(&out[(mbase + j) * NB + b], acc[j] * SCALE);
    }
}

extern "C" void kernel_launch(void* const* d_in, const int* in_sizes, int n_in,
                              void* d_out, int out_size, void* d_ws, size_t ws_size,
                              hipStream_t stream) {
    const float* data = (const float*)d_in[0];   // [16, 32768] f32
    // d_in[1] is `dim` (= -1), layout already matches; ignored.
    const float* cX = (const float*)d_in[2];     // [256] f32
    float* out = (float*)d_out;                  // [256, 16] f32

    (void)in_sizes; (void)n_in; (void)d_ws; (void)ws_size;

    kde_zero<<<out_size / 256, 256, 0, stream>>>(out);
    kde_wmma<<<NB * SPLITS, WG_THREADS, 0, stream>>>(data, cX, out);
}
